// FIRE_RED_DECODER_55868934586423
// MI455X (gfx1250) — matrix operand order repaired and checked by reading, not compiled, not run on hardware
//
#include <hip/hip_runtime.h>

// ---------------- shapes (static per reference) ----------------
namespace cfg {
constexpr int L    = 16;
constexpr int H    = 20;
constexpr int DH   = 64;
constexpr int D    = 1280;
constexpr int DFF  = 5120;
constexpr int VOC  = 8000;
constexpr int ENC  = 1024;
constexpr int T    = 448;
constexpr int HIST = 64;
constexpr int KV   = HIST + T;            // 512
constexpr long long TD = (long long)T * D;    // 573440
}

typedef __attribute__((ext_vector_type(16))) __bf16 bf16x16;
typedef __attribute__((ext_vector_type(8)))  float  f32x8;

// ---------------- block reductions (wave32) ----------------
__device__ __forceinline__ float block_sum(float v) {
    __shared__ float sh[8];
    __shared__ float outv;
    #pragma unroll
    for (int o = 16; o > 0; o >>= 1) v += __shfl_xor(v, o, 32);
    const int lane = threadIdx.x & 31, w = threadIdx.x >> 5, nw = blockDim.x >> 5;
    if (lane == 0) sh[w] = v;
    __syncthreads();
    if (threadIdx.x == 0) {
        float s = sh[0];
        for (int i = 1; i < nw; ++i) s += sh[i];
        outv = s;
    }
    __syncthreads();
    return outv;
}

__device__ __forceinline__ float block_max(float v) {
    __shared__ float sh[8];
    __shared__ float outv;
    #pragma unroll
    for (int o = 16; o > 0; o >>= 1) v = fmaxf(v, __shfl_xor(v, o, 32));
    const int lane = threadIdx.x & 31, w = threadIdx.x >> 5, nw = blockDim.x >> 5;
    if (lane == 0) sh[w] = v;
    __syncthreads();
    if (threadIdx.x == 0) {
        float s = sh[0];
        for (int i = 1; i < nw; ++i) s = fmaxf(s, sh[i]);
        outv = s;
    }
    __syncthreads();
    return outv;
}

// ---------------- generic batched WMMA GEMM ----------------
// C[m*ldcM + n*ldcN] = act( sum_k A[m*lda+k] * B[k*ldb+n] + bias[n] ) (+resid)
// flags: 1=bias, 2=relu, 4=residual. Requires M%32==0, N%64==0, K%32==0.
// Per wave: 32x64 C macro-tile = 2 A frags x 4 B frags = 8 WMMAs / 32-K step.
__global__ __launch_bounds__(128)
void wmma_gemm_bf16(const float* __restrict__ A, const float* __restrict__ B,
                    float* __restrict__ C, const float* __restrict__ bias,
                    const float* __restrict__ resid,
                    int M, int N, int K, int lda, int ldb, int ldcM, int ldcN,
                    long long sA, long long sB, long long sC, int sBias, int flags)
{
    const int batch = blockIdx.z;
    A += (long long)batch * sA;
    B += (long long)batch * sB;
    C += (long long)batch * sC;
    if (bias)  bias  += (long long)batch * sBias;
    if (resid) resid += (long long)batch * sC;

    const int wave  = threadIdx.x >> 5;
    const int lane  = threadIdx.x & 31;
    const int mTile = blockIdx.y * 32;
    const int nBase = (blockIdx.x * 4 + wave) * 64;
    if (nBase >= N || mTile >= M) return;

    const int half = lane >> 4;        // 0: lanes 0-15, 1: lanes 16-31
    const int ml   = lane & 15;
    const int kbA  = half * 16;        // A frag K-base (contiguous 16 K per half)
    const int kbB  = half * 16;        // B frag K-base

    f32x8 acc0[4] = {};
    f32x8 acc1[4] = {};

    const float* Arow0 = A + (long long)(mTile + ml) * lda;
    const float* Arow1 = Arow0 + (long long)16 * lda;

    for (int k0 = 0; k0 < K; k0 += 32) {
        // ---- two A fragments: rows [mTile, mTile+16) and [mTile+16, mTile+32) ----
        bf16x16 a0, a1;
        {
            const float* Ar0 = Arow0 + k0 + kbA;
            const float* Ar1 = Arow1 + k0 + kbA;
            #pragma unroll
            for (int j = 0; j < 4; ++j) {
                float2 p0 = *(const float2*)(Ar0 + 2 * j);
                float2 p1 = *(const float2*)(Ar0 + 8 + 2 * j);
                a0[2 * j]     = (__bf16)p0.x;  a0[2 * j + 1]     = (__bf16)p0.y;
                a0[8 + 2 * j] = (__bf16)p1.x;  a0[8 + 2 * j + 1] = (__bf16)p1.y;
                float2 q0 = *(const float2*)(Ar1 + 2 * j);
                float2 q1 = *(const float2*)(Ar1 + 8 + 2 * j);
                a1[2 * j]     = (__bf16)q0.x;  a1[2 * j + 1]     = (__bf16)q0.y;
                a1[8 + 2 * j] = (__bf16)q1.x;  a1[8 + 2 * j + 1] = (__bf16)q1.y;
            }
        }
        // ---- 4 B fragments (32x16 each), each reused by two WMMAs ----
        const float* Bc = B + (long long)(k0 + kbB) * ldb + nBase + ml;
        if (k0 + 32 < K) __builtin_prefetch(Bc + 32 * ldb, 0, 3);
        #pragma unroll
        for (int t = 0; t < 4; ++t) {
            const float* Bt = Bc + 16 * t;
            bf16x16 b;
            #pragma unroll
            for (int j = 0; j < 8; ++j) {
                b[2 * j]     = (__bf16)Bt[(2 * j) * (long long)ldb];
                b[2 * j + 1] = (__bf16)Bt[(2 * j + 1) * (long long)ldb];
            }
#if defined(__gfx1250__)
            acc0[t] = __builtin_amdgcn_wmma_f32_16x16x32_bf16(
                false, a0, false, b, (short)0, acc0[t], false, false);
            acc1[t] = __builtin_amdgcn_wmma_f32_16x16x32_bf16(
                false, a1, false, b, (short)0, acc1[t], false, false);
#endif
        }
    }

    // ---- epilogue: C/D 16x16 f32 layout ----
    const int nloc = lane & 15;
    const int mOff = half * 8;
    #pragma unroll
    for (int mi = 0; mi < 2; ++mi) {
        const int mBase = mTile + 16 * mi + mOff;
        #pragma unroll
        for (int t = 0; t < 4; ++t) {
            const int n = nBase + 16 * t + nloc;
            const float bb = (flags & 1) ? bias[n] : 0.f;
            #pragma unroll
            for (int r = 0; r < 8; ++r) {
                const int m = mBase + r;
                float v = (mi == 0 ? acc0[t][r] : acc1[t][r]) + bb;
                const long long ci = (long long)m * ldcM + (long long)n * ldcN;
                if (flags & 4) v += resid[ci];
                if (flags & 2) v = v > 0.f ? v : 0.f;
                C[ci] = v;
            }
        }
    }
}

// ---------------- LayerNorm (row-wise over D) ----------------
__global__ void ln_kernel(const float* __restrict__ X, float* __restrict__ Y,
                          const float* __restrict__ W, const float* __restrict__ B, int D)
{
    const int row = blockIdx.x;
    const float* x = X + (long long)row * D;
    float* y = Y + (long long)row * D;
    float s = 0.f;
    for (int i = threadIdx.x; i < D; i += blockDim.x) s += x[i];
    const float mean = block_sum(s) / D;
    float v = 0.f;
    for (int i = threadIdx.x; i < D; i += blockDim.x) { float d = x[i] - mean; v += d * d; }
    const float var = block_sum(v) / D;
    const float inv = rsqrtf(var + 1e-5f);
    for (int i = threadIdx.x; i < D; i += blockDim.x)
        y[i] = (x[i] - mean) * inv * W[i] + B[i];
}

// ---------------- softmax (optionally causal-masked) ----------------
// S is [H*T, len] row-major, row = head*T + t; mask: +(-128*mask_flag) where j > t
__global__ void softmax_kernel(float* __restrict__ S, const float* __restrict__ mask_flag,
                               int T, int len, int causal)
{
    const int row = blockIdx.x;
    const int t = row % T;
    float* sr = S + (long long)row * len;
    const float mf = causal ? (-128.f * mask_flag[0]) : 0.f;
    const int cnt = len >> 8;   // len/256, <= 4
    float vals[4];
    float mx = -3.4e38f;
    for (int i = 0; i < cnt; ++i) {
        const int j = threadIdx.x + (i << 8);
        float v = sr[j];
        if (causal && j > t) v += mf;
        vals[i] = v;
        mx = fmaxf(mx, v);
    }
    mx = block_max(mx);
    float ssum = 0.f;
    for (int i = 0; i < cnt; ++i) { vals[i] = __expf(vals[i] - mx); ssum += vals[i]; }
    ssum = block_sum(ssum);
    const float inv = 1.f / ssum;
    for (int i = 0; i < cnt; ++i) sr[threadIdx.x + (i << 8)] = vals[i] * inv;
}

// ---------------- embedding + positional ----------------
__global__ void embed_kernel(const int* __restrict__ ids, const float* __restrict__ emb,
                             const float* __restrict__ pe, float* __restrict__ h,
                             int T, int D, int hist)
{
    const long long i = (long long)blockIdx.x * blockDim.x + threadIdx.x;
    if (i >= (long long)T * D) return;
    const int t = (int)(i / D), d = (int)(i % D);
    h[i] = emb[(long long)ids[t] * D + d] + pe[(long long)(hist + t) * D + d];
}

// ---------------- KV-cache history copies into outputs ----------------
// new_k[l,h,e, j<HIST] = cache[l,h,e,j]  (row len KV vs HIST)
__global__ void copy_kcache(const float* __restrict__ src, float* __restrict__ dst,
                            long long n, int HIST, int KV)
{
    const long long i = (long long)blockIdx.x * blockDim.x + threadIdx.x;
    if (i >= n) return;
    const long long j = i % HIST, rest = i / HIST;
    dst[rest * KV + j] = src[i];
}
// new_v[l,h, r<HIST, e] = cache[l,h,r,e]
__global__ void copy_vcache(const float* __restrict__ src, float* __restrict__ dst,
                            long long n, int HIST, int KV, int DH)
{
    const long long i = (long long)blockIdx.x * blockDim.x + threadIdx.x;
    if (i >= n) return;
    const long long inner = i % ((long long)HIST * DH);
    const long long lh    = i / ((long long)HIST * DH);
    dst[lh * (long long)KV * DH + inner] = src[i];
}

// ---------------- final projection row (M=1: plain dot, WMMA wasteful) ----------------
__global__ void logits_kernel(const float* __restrict__ hn, const float* __restrict__ pw,
                              const float* __restrict__ pb, float* __restrict__ out,
                              int D, int V, int lastRow)
{
    __shared__ float xs[1280];
    for (int i = threadIdx.x; i < D; i += blockDim.x)
        xs[i] = hn[(long long)lastRow * D + i];
    __syncthreads();
    const int n = blockIdx.x * blockDim.x + threadIdx.x;
    if (n >= V) return;
    float s = pb[n];
    for (int d = 0; d < D; ++d) s += xs[d] * pw[(long long)d * V + n];
    out[n] = s;
}

__global__ void argmax_kernel(const float* __restrict__ logits, int V,
                              float* __restrict__ meta, int kvlen)
{
    float best = -3.4e38f; int bidx = 0x7fffffff;
    for (int i = threadIdx.x; i < V; i += blockDim.x) {
        const float v = logits[i];
        if (v > best || (v == best && i < bidx)) { best = v; bidx = i; }
    }
    __shared__ float bv[256];
    __shared__ int   bi[256];
    bv[threadIdx.x] = best; bi[threadIdx.x] = bidx;
    __syncthreads();
    for (int s = 128; s > 0; s >>= 1) {
        if ((int)threadIdx.x < s) {
            if (bv[threadIdx.x + s] > bv[threadIdx.x] ||
                (bv[threadIdx.x + s] == bv[threadIdx.x] && bi[threadIdx.x + s] < bi[threadIdx.x])) {
                bv[threadIdx.x] = bv[threadIdx.x + s];
                bi[threadIdx.x] = bi[threadIdx.x + s];
            }
        }
        __syncthreads();
    }
    if (threadIdx.x == 0) { meta[0] = (float)kvlen; meta[1] = (float)bi[0]; }
}

// ---------------- host orchestration ----------------
static inline void gemm(hipStream_t s,
                        const float* A, const float* B, float* C,
                        const float* bias, const float* resid,
                        int M, int N, int K, int lda, int ldb, int ldcM, int ldcN,
                        long long sA, long long sB, long long sC, int sBias,
                        int batch, int flags)
{
    dim3 grid((N + 255) / 256, (M + 31) / 32, batch);
    wmma_gemm_bf16<<<grid, dim3(128), 0, s>>>(A, B, C, bias, resid,
        M, N, K, lda, ldb, ldcM, ldcN, sA, sB, sC, sBias, flags);
}

extern "C" void kernel_launch(void* const* d_in, const int* in_sizes, int n_in,
                              void* d_out, int out_size, void* d_ws, size_t ws_size,
                              hipStream_t stream)
{
    using namespace cfg;
    (void)in_sizes; (void)n_in; (void)out_size; (void)ws_size;

    const float* skc  = (const float*)d_in[0];
    const float* svc  = (const float*)d_in[1];
    const float* ck   = (const float*)d_in[2];
    const float* cv   = (const float*)d_in[3];
    const int*   ids  = (const int*)  d_in[4];
    const float* mflg = (const float*)d_in[7];
    const float* emb  = (const float*)d_in[8];
    const float* pe   = (const float*)d_in[9];
    const float* lnsw = (const float*)d_in[10];
    const float* lnsb = (const float*)d_in[11];
    const float* wq   = (const float*)d_in[12];
    const float* bq   = (const float*)d_in[13];
    const float* wk   = (const float*)d_in[14];
    const float* wv   = (const float*)d_in[15];
    const float* bv   = (const float*)d_in[16];
    const float* wo   = (const float*)d_in[17];
    const float* bo   = (const float*)d_in[18];
    const float* lncw = (const float*)d_in[19];
    const float* lncb = (const float*)d_in[20];
    const float* cwq  = (const float*)d_in[21];
    const float* cbq  = (const float*)d_in[22];
    const float* cwo  = (const float*)d_in[23];
    const float* cbo  = (const float*)d_in[24];
    const float* lnmw = (const float*)d_in[25];
    const float* lnmb = (const float*)d_in[26];
    const float* w1   = (const float*)d_in[27];
    const float* b1   = (const float*)d_in[28];
    const float* w2   = (const float*)d_in[29];
    const float* b2   = (const float*)d_in[30];
    const float* lnow = (const float*)d_in[31];
    const float* lnob = (const float*)d_in[32];
    const float* pw   = (const float*)d_in[33];
    const float* pb   = (const float*)d_in[34];

    float* outK  = (float*)d_out;                                  // [L,H,DH,KV]
    float* outV  = outK + (long long)L * H * DH * KV;              // [L,H,KV,DH]
    float* outMeta = outV + (long long)L * H * KV * DH;            // [kv_len, max_idx]

    float* ws    = (float*)d_ws;
    float* bufH  = ws;               ws += TD;     // h
    float* bufHN = ws;               ws += TD;     // LN scratch
    float* bufT1 = ws;               ws += TD;     // q_all / cq_all   [T, H*DH]
    float* bufT2 = ws;               ws += TD;     // o_all / co_all   [T, H*DH]
    float* bufHA = ws;               ws += TD;     // h_attn
    float* bufHC = ws;               ws += TD;     // h_cross
    float* bufFF = ws;               ws += (long long)T * DFF;      // MLP hidden
    float* bufSC = ws;               ws += (long long)H * T * ENC;  // scores (self/cross)
    float* bufLG = ws;               ws += VOC;                     // logits

    // embedding + PE
    embed_kernel<<<(unsigned)((TD + 255) / 256), 256, 0, stream>>>(ids, emb, pe, bufH, T, D, HIST);

    // history halves of new_k / new_v
    {
        const long long nk = (long long)L * H * DH * HIST;
        copy_kcache<<<(unsigned)((nk + 255) / 256), 256, 0, stream>>>(skc, outK, nk, HIST, KV);
        const long long nv = (long long)L * H * HIST * DH;
        copy_vcache<<<(unsigned)((nv + 255) / 256), 256, 0, stream>>>(svc, outV, nv, HIST, KV, DH);
    }

    for (int l = 0; l < L; ++l) {
        const float* wq_l  = wq  + (long long)l * H * D * DH;
        const float* bq_l  = bq  + (long long)l * H * DH;
        const float* wk_l  = wk  + (long long)l * H * D * DH;
        const float* wv_l  = wv  + (long long)l * H * D * DH;
        const float* bv_l  = bv  + (long long)l * H * DH;
        const float* wo_l  = wo  + (long long)l * H * DH * D;   // [H*DH, D] row-major
        const float* bo_l  = bo  + (long long)l * D;
        const float* ck_l  = ck  + (long long)l * H * DH * ENC;
        const float* cv_l  = cv  + (long long)l * H * ENC * DH;
        const float* cwq_l = cwq + (long long)l * H * D * DH;
        const float* cbq_l = cbq + (long long)l * H * DH;
        const float* cwo_l = cwo + (long long)l * H * DH * D;
        const float* cbo_l = cbo + (long long)l * D;
        const float* w1_l  = w1  + (long long)l * D * DFF;
        const float* b1_l  = b1  + (long long)l * DFF;
        const float* w2_l  = w2  + (long long)l * DFF * D;
        const float* b2_l  = b2  + (long long)l * D;
        float* kOut_l = outK + (long long)l * H * DH * KV;      // per-head [DH, KV]
        float* vOut_l = outV + (long long)l * H * KV * DH;      // per-head [KV, DH]

        // ---- self attention ----
        ln_kernel<<<T, 256, 0, stream>>>(bufH, bufHN, lnsw + (long long)l * D, lnsb + (long long)l * D, D);
        // q = hn @ wq[h] + bq[h]      -> bufT1[:, h*64:(h+1)*64]
        gemm(stream, bufHN, wq_l, bufT1, bq_l, nullptr,
             T, DH, D, D, DH, D, 1, 0, (long long)D * DH, DH, DH, H, 1);
        // k = hn @ wk[h], written transposed into new_k[l,h,:,HIST:]
        gemm(stream, bufHN, wk_l, kOut_l + HIST, nullptr, nullptr,
             T, DH, D, D, DH, 1, KV, 0, (long long)D * DH, (long long)DH * KV, 0, H, 0);
        // v = hn @ wv[h] + bv[h], written into new_v[l,h,HIST:,:]
        gemm(stream, bufHN, wv_l, vOut_l + (long long)HIST * DH, bv_l, nullptr,
             T, DH, D, D, DH, DH, 1, 0, (long long)D * DH, (long long)KV * DH, DH, H, 1);
        // scores[h] = q[h] @ K[h]     (K from d_out, includes history)
        gemm(stream, bufT1, kOut_l, bufSC, nullptr, nullptr,
             T, KV, DH, D, KV, KV, 1, DH, (long long)DH * KV, (long long)T * KV, 0, H, 0);
        softmax_kernel<<<H * T, 256, 0, stream>>>(bufSC, mflg, T, KV, 1);
        // o[h] = attn[h] @ V[h]       -> bufT2[:, h*64:(h+1)*64]
        gemm(stream, bufSC, vOut_l, bufT2, nullptr, nullptr,
             T, DH, KV, KV, DH, D, 1, (long long)T * KV, (long long)KV * DH, DH, 0, H, 0);
        // out-proj over all heads + bo + residual(h)
        gemm(stream, bufT2, wo_l, bufHA, bo_l, bufH,
             T, D, D, D, D, D, 1, 0, 0, 0, 0, 1, 1 | 4);

        // ---- cross attention ----
        ln_kernel<<<T, 256, 0, stream>>>(bufHA, bufHN, lncw + (long long)l * D, lncb + (long long)l * D, D);
        gemm(stream, bufHN, cwq_l, bufT1, cbq_l, nullptr,
             T, DH, D, D, DH, D, 1, 0, (long long)D * DH, DH, DH, H, 1);
        gemm(stream, bufT1, ck_l, bufSC, nullptr, nullptr,
             T, ENC, DH, D, ENC, ENC, 1, DH, (long long)DH * ENC, (long long)T * ENC, 0, H, 0);
        softmax_kernel<<<H * T, 256, 0, stream>>>(bufSC, mflg, T, ENC, 0);
        gemm(stream, bufSC, cv_l, bufT2, nullptr, nullptr,
             T, DH, ENC, ENC, DH, D, 1, (long long)T * ENC, (long long)ENC * DH, DH, 0, H, 0);
        gemm(stream, bufT2, cwo_l, bufHC, cbo_l, bufHA,
             T, D, D, D, D, D, 1, 0, 0, 0, 0, 1, 1 | 4);

        // ---- MLP ----
        ln_kernel<<<T, 256, 0, stream>>>(bufHC, bufHN, lnmw + (long long)l * D, lnmb + (long long)l * D, D);
        gemm(stream, bufHN, w1_l, bufFF, b1_l, nullptr,
             T, DFF, D, D, DFF, DFF, 1, 0, 0, 0, 0, 1, 1 | 2);
        gemm(stream, bufFF, w2_l, bufH, b2_l, bufHC,
             T, D, DFF, DFF, D, D, 1, 0, 0, 0, 0, 1, 1 | 4);
    }

    // final LN + projection of last token + argmax
    ln_kernel<<<T, 256, 0, stream>>>(bufH, bufHN, lnow, lnob, D);
    logits_kernel<<<(VOC + 255) / 256, 256, 0, stream>>>(bufHN, pw, pb, bufLG, D, VOC, T - 1);
    argmax_kernel<<<1, 256, 0, stream>>>(bufLG, VOC, outMeta, HIST + T);
}
